// GAT_65017214927221
// MI455X (gfx1250) — matrix operand order repaired and checked by reading, not compiled, not run on hardware
//
#include <hip/hip_runtime.h>

// ---------------------------------------------------------------------------
// Problem constants (from reference)
// ---------------------------------------------------------------------------
#define NN      10000      // nodes
#define EE      160000     // edges (before self loops)
#define EP      (EE + NN)  // edges incl. self loops
#define NHEAD   8
#define BGRAPH  16
#define NEGSLOPE 0.2f

typedef __attribute__((ext_vector_type(16))) __bf16 bf16x16;
typedef __attribute__((ext_vector_type(8)))  __bf16 bf16x8;
typedef __attribute__((ext_vector_type(8)))  float  f32x8;

__device__ __forceinline__ unsigned short f32_to_bf16(float f) {
    unsigned u = __float_as_uint(f);
    unsigned r = u + 0x7FFFu + ((u >> 16) & 1u);   // round to nearest even
    return (unsigned short)(r >> 16);
}

// ---------------------------------------------------------------------------
// Utility kernels
// ---------------------------------------------------------------------------
__global__ void fill_f32(float* p, float v, size_t n) {
    size_t i = (size_t)blockIdx.x * blockDim.x + threadIdx.x;
    if (i < n) p[i] = v;
}

__global__ void cvt_bf16_kernel(const float* __restrict__ in, unsigned short* __restrict__ out, size_t n) {
    size_t i = (size_t)blockIdx.x * blockDim.x + threadIdx.x;
    if (i < n) out[i] = f32_to_bf16(in[i]);
}

// W is [K, Nout] row-major; WT is [Nout, K] row-major (bf16)
__global__ void transpose_bf16_kernel(const float* __restrict__ W, unsigned short* __restrict__ WT,
                                      int K, int Nout) {
    size_t i = (size_t)blockIdx.x * blockDim.x + threadIdx.x;
    if (i >= (size_t)K * Nout) return;
    int nn = (int)(i / K);
    int k  = (int)(i % K);
    WT[i] = f32_to_bf16(W[(size_t)k * Nout + nn]);
}

// ---------------------------------------------------------------------------
// WMMA bf16 GEMM:  C[M,Nn] (f32) = A[M,K] (bf16 row-major) * BT[Nn,K] (bf16)
// Block: 128 threads = 4 waves; wave -> 16x64 tile; block -> 16x256.
// grid.x = M/16, grid.y = Nn/256.  K multiple of 32.
// ---------------------------------------------------------------------------
__global__ void gemm_wmma_bf16(const unsigned short* __restrict__ A,
                               const unsigned short* __restrict__ BT,
                               float* __restrict__ C, int M, int Nn, int K) {
    const int wave = threadIdx.x >> 5;
    const int lane = threadIdx.x & 31;
    const int hi   = lane >> 4;        // 0: lanes 0-15, 1: lanes 16-31
    const int l15  = lane & 15;

    const int mBase = blockIdx.x * 16;
    const int nBase = (blockIdx.y * 4 + wave) * 64;

    f32x8 acc0 = {}, acc1 = {}, acc2 = {}, acc3 = {};

    const unsigned short* Arow = A + (size_t)(mBase + l15) * K + (hi << 4);
    const unsigned short* Bcol = BT + (size_t)(nBase + l15) * K + (hi << 4);

    for (int k0 = 0; k0 < K; k0 += 32) {
        // A tile: lanes 0-15 hold row M=l15, K = k0..k0+15;
        //         lanes 16-31 K = k0+16..k0+31 (contiguous, matches B).
        bf16x8 alo = *(const bf16x8*)(const void*)(Arow + k0);
        bf16x8 ahi = *(const bf16x8*)(const void*)(Arow + k0 + 8);
        bf16x16 a = __builtin_shufflevector(alo, ahi,
                        0,1,2,3,4,5,6,7,8,9,10,11,12,13,14,15);

        // B tile: 16 contiguous K values of this lane's column (BT row-major)
        bf16x16 b0 = *(const bf16x16*)(const void*)(Bcol + k0);
        bf16x16 b1 = *(const bf16x16*)(const void*)(Bcol + k0 + (size_t)16 * K);
        bf16x16 b2 = *(const bf16x16*)(const void*)(Bcol + k0 + (size_t)32 * K);
        bf16x16 b3 = *(const bf16x16*)(const void*)(Bcol + k0 + (size_t)48 * K);

        acc0 = __builtin_amdgcn_wmma_f32_16x16x32_bf16(false, a, false, b0, (short)0, acc0, false, false);
        acc1 = __builtin_amdgcn_wmma_f32_16x16x32_bf16(false, a, false, b1, (short)0, acc1, false, false);
        acc2 = __builtin_amdgcn_wmma_f32_16x16x32_bf16(false, a, false, b2, (short)0, acc2, false, false);
        acc3 = __builtin_amdgcn_wmma_f32_16x16x32_bf16(false, a, false, b3, (short)0, acc3, false, false);
    }

    // C layout: VGPR r -> row mBase + r + 8*hi, col nBase + l15 (+16 per acc)
    const int row0 = mBase + (hi << 3);
    for (int r = 0; r < 8; ++r) {
        float* Cp = C + (size_t)(row0 + r) * Nn + nBase + l15;
        Cp[0]  = acc0[r];
        Cp[16] = acc1[r];
        Cp[32] = acc2[r];
        Cp[48] = acc3[r];
    }
}

// ---------------------------------------------------------------------------
// alpha_src / alpha_dst : per (node, head) dot products. 8 waves = 8 heads.
// ---------------------------------------------------------------------------
__global__ void alpha_kernel(const float* __restrict__ h,
                             const float* __restrict__ attS,
                             const float* __restrict__ attD,
                             float* __restrict__ aS, float* __restrict__ aD, int C) {
    const int n    = blockIdx.x;
    const int head = threadIdx.x >> 5;
    const int lane = threadIdx.x & 31;
    const float* hp = h + (size_t)n * (NHEAD * C) + (size_t)head * C;
    const float* sp = attS + (size_t)head * C;
    const float* dp = attD + (size_t)head * C;
    float s0 = 0.f, d0 = 0.f;
    for (int c = lane; c < C; c += 32) {
        float v = hp[c];
        s0 += v * sp[c];
        d0 += v * dp[c];
    }
    for (int off = 16; off; off >>= 1) {
        s0 += __shfl_down(s0, off);
        d0 += __shfl_down(d0, off);
    }
    if (lane == 0) {
        aS[n * NHEAD + head] = s0;
        aD[n * NHEAD + head] = d0;
    }
}

__device__ __forceinline__ void edge_nodes(const int* __restrict__ ei, int e, int& s, int& d) {
    if (e < EE) { s = ei[e]; d = ei[EE + e]; }
    else        { s = d = e - EE; }           // self loop
}

// leaky-relu logit + segment max (int-trick float atomic max; m init = -inf-ish)
__global__ void edge_pass1(const int* __restrict__ ei,
                           const float* __restrict__ aS, const float* __restrict__ aD,
                           float* __restrict__ eb, float* __restrict__ m) {
    int idx = blockIdx.x * blockDim.x + threadIdx.x;
    if (idx >= EP * NHEAD) return;
    int e = idx >> 3, hd = idx & 7;
    int s, d; edge_nodes(ei, e, s, d);
    float v = aS[s * NHEAD + hd] + aD[d * NHEAD + hd];
    v = (v > 0.f) ? v : NEGSLOPE * v;
    eb[idx] = v;
    float* addr = m + d * NHEAD + hd;
    if (v >= 0.f) atomicMax((int*)addr, __float_as_int(v));
    else          atomicMin((unsigned int*)addr, __float_as_uint(v));
}

// exp(e - m[dst]) + segment sum
__global__ void edge_pass2(const int* __restrict__ ei,
                           float* __restrict__ eb,
                           const float* __restrict__ m, float* __restrict__ denom) {
    int idx = blockIdx.x * blockDim.x + threadIdx.x;
    if (idx >= EP * NHEAD) return;
    int e = idx >> 3, hd = idx & 7;
    int s, d; edge_nodes(ei, e, s, d);
    (void)s;
    float ex = __expf(eb[idx] - m[d * NHEAD + hd]);
    eb[idx] = ex;
    atomicAdd(denom + d * NHEAD + hd, ex);
}

// weighted scatter: acc[dst] += (ex/denom[dst]) * h[src]. One block per edge.
// HC = heads*C; each thread covers HC/256 consecutive channels (same head).
__global__ void edge_pass3(const int* __restrict__ ei,
                           const float* __restrict__ eb, const float* __restrict__ denom,
                           const float* __restrict__ h, float* __restrict__ acc,
                           int HC, int C) {
    const int e = blockIdx.x;
    int s, d; edge_nodes(ei, e, s, d);
    const int per = HC >> 8;                 // 4 (conv1) or 2 (conv2)
    const int j0  = threadIdx.x * per;
    const int hd  = j0 / C;
    const float coef = eb[(size_t)e * NHEAD + hd] / denom[d * NHEAD + hd];
    const float* hs = h + (size_t)s * HC + j0;
    float* ap = acc + (size_t)d * HC + j0;
    for (int j = 0; j < per; ++j) atomicAdd(ap + j, coef * hs[j]);
}

// bias + ELU -> bf16 (feeds next GEMM's A operand)
__global__ void finalize_bf16(const float* __restrict__ acc, const float* __restrict__ bias,
                              unsigned short* __restrict__ outb, int HC) {
    const int n = blockIdx.x;
    for (int j = threadIdx.x; j < HC; j += blockDim.x) {
        float v = acc[(size_t)n * HC + j] + bias[j];
        v = (v > 0.f) ? v : (__expf(v) - 1.0f);
        outb[(size_t)n * HC + j] = f32_to_bf16(v);
    }
}

// bias + ELU in place (f32)
__global__ void finalize_f32(float* __restrict__ acc, const float* __restrict__ bias, int HC) {
    const int n = blockIdx.x;
    for (int j = threadIdx.x; j < HC; j += blockDim.x) {
        float v = acc[(size_t)n * HC + j] + bias[j];
        acc[(size_t)n * HC + j] = (v > 0.f) ? v : (__expf(v) - 1.0f);
    }
}

// global mean-pool accumulation
__global__ void pool_kernel(const float* __restrict__ h2, const int* __restrict__ batch,
                            float* __restrict__ pooled, float* __restrict__ counts) {
    const int n = blockIdx.x;
    const int b = batch[n];
    for (int j = threadIdx.x; j < 512; j += blockDim.x)
        atomicAdd(pooled + b * 512 + j, h2[(size_t)n * 512 + j]);
    if (threadIdx.x == 0) atomicAdd(counts + b, 1.0f);
}

// mean + fc1(ELU) + fc2 -> out[16,10]; single block, 512 threads
__global__ void fc_head(const float* __restrict__ pooled, const float* __restrict__ counts,
                        const float* __restrict__ w1, const float* __restrict__ b1,
                        const float* __restrict__ w2, const float* __restrict__ b2,
                        float* __restrict__ out) {
    __shared__ float g1[BGRAPH * 32];
    const int tid = threadIdx.x;
    {
        const int row = tid >> 5, col = tid & 31;     // 16x32 = 512 outputs
        const float inv = 1.0f / fmaxf(counts[row], 1.0f);
        float a = b1[col];
        for (int k = 0; k < 512; ++k)
            a += pooled[row * 512 + k] * inv * w1[k * 32 + col];
        g1[row * 32 + col] = (a > 0.f) ? a : (__expf(a) - 1.0f);
    }
    __syncthreads();
    if (tid < BGRAPH * 10) {
        const int row = tid / 10, col = tid % 10;
        float a = b2[col];
        for (int k = 0; k < 32; ++k) a += g1[row * 32 + k] * w2[k * 10 + col];
        out[row * 10 + col] = a;
    }
}

// ---------------------------------------------------------------------------
// Host-side orchestration
// ---------------------------------------------------------------------------
static inline size_t align256(size_t x) { return (x + 255) & ~(size_t)255; }
static inline unsigned nblk(size_t n, int t) { return (unsigned)((n + (size_t)t - 1) / (size_t)t); }

extern "C" void kernel_launch(void* const* d_in, const int* in_sizes, int n_in,
                              void* d_out, int out_size, void* d_ws, size_t ws_size,
                              hipStream_t stream) {
    (void)in_sizes; (void)n_in; (void)out_size; (void)ws_size;

    const float* x     = (const float*)d_in[0];
    const int*   ei    = (const int*)  d_in[1];
    const int*   batch = (const int*)  d_in[2];
    const float* W1    = (const float*)d_in[3];
    const float* at_s1 = (const float*)d_in[4];
    const float* at_d1 = (const float*)d_in[5];
    const float* b1    = (const float*)d_in[6];
    const float* W2    = (const float*)d_in[7];
    const float* at_s2 = (const float*)d_in[8];
    const float* at_d2 = (const float*)d_in[9];
    const float* b2    = (const float*)d_in[10];
    const float* fc1w  = (const float*)d_in[11];
    const float* fc1b  = (const float*)d_in[12];
    const float* fc2w  = (const float*)d_in[13];
    const float* fc2b  = (const float*)d_in[14];
    float* out = (float*)d_out;

    // ---- workspace layout (phase-aliased, ~93 MB total) ----
    char* ws = (char*)d_ws;
    size_t off = 0;
    const size_t oA1  = off; off += align256((size_t)NN * 128 * 2);     // x bf16
    const size_t oW1T = off; off += align256((size_t)1024 * 128 * 2);   // W1^T bf16
    const size_t oR1  = off; off += align256((size_t)NN * 1024 * 4);    // h1lin f32 | later: A2 bf16 + acc2 f32
    const size_t oR2  = off; off += align256((size_t)NN * 1024 * 4);    // acc1 f32  | later: h2lin f32
    const size_t oW2T = off; off += align256((size_t)512 * 1024 * 2);   // W2^T bf16
    const size_t oAS  = off; off += align256((size_t)NN * NHEAD * 4);
    const size_t oAD  = off; off += align256((size_t)NN * NHEAD * 4);
    const size_t oM   = off; off += align256((size_t)NN * NHEAD * 4);
    const size_t oDN  = off; off += align256((size_t)NN * NHEAD * 4);
    const size_t oEB  = off; off += align256((size_t)EP * NHEAD * 4);
    const size_t oPO  = off; off += align256((size_t)BGRAPH * 512 * 4);
    const size_t oCT  = off; off += align256((size_t)BGRAPH * 4);

    unsigned short* A1   = (unsigned short*)(ws + oA1);
    unsigned short* W1T  = (unsigned short*)(ws + oW1T);
    float*          h1   = (float*)(ws + oR1);
    unsigned short* A2   = (unsigned short*)(ws + oR1);                    // aliases h1 (dead by then)
    float*          acc2 = (float*)(ws + oR1 + (size_t)NN * 1024 * 2);    // second half of R1
    float*          acc1 = (float*)(ws + oR2);
    float*          h2   = (float*)(ws + oR2);                             // aliases acc1 (dead by then)
    unsigned short* W2T  = (unsigned short*)(ws + oW2T);
    float* aS   = (float*)(ws + oAS);
    float* aD   = (float*)(ws + oAD);
    float* mBuf = (float*)(ws + oM);
    float* dnB  = (float*)(ws + oDN);
    float* eb   = (float*)(ws + oEB);
    float* pooled = (float*)(ws + oPO);
    float* counts = (float*)(ws + oCT);

    // ---- convert inputs to bf16 WMMA layouts ----
    cvt_bf16_kernel<<<nblk((size_t)NN * 128, 256), 256, 0, stream>>>(x, A1, (size_t)NN * 128);
    transpose_bf16_kernel<<<nblk((size_t)128 * 1024, 256), 256, 0, stream>>>(W1, W1T, 128, 1024);
    transpose_bf16_kernel<<<nblk((size_t)1024 * 512, 256), 256, 0, stream>>>(W2, W2T, 1024, 512);

    // ================= conv1 =================
    gemm_wmma_bf16<<<dim3(NN / 16, 1024 / 256), 128, 0, stream>>>(A1, W1T, h1, NN, 1024, 128);
    alpha_kernel<<<NN, 256, 0, stream>>>(h1, at_s1, at_d1, aS, aD, 128);

    fill_f32<<<nblk((size_t)NN * NHEAD, 256), 256, 0, stream>>>(mBuf, -3.0e38f, (size_t)NN * NHEAD);
    fill_f32<<<nblk((size_t)NN * NHEAD, 256), 256, 0, stream>>>(dnB, 0.f, (size_t)NN * NHEAD);
    fill_f32<<<nblk((size_t)NN * 1024, 256), 256, 0, stream>>>(acc1, 0.f, (size_t)NN * 1024);

    edge_pass1<<<nblk((size_t)EP * NHEAD, 256), 256, 0, stream>>>(ei, aS, aD, eb, mBuf);
    edge_pass2<<<nblk((size_t)EP * NHEAD, 256), 256, 0, stream>>>(ei, eb, mBuf, dnB);
    edge_pass3<<<EP, 256, 0, stream>>>(ei, eb, dnB, h1, acc1, 1024, 128);
    finalize_bf16<<<NN, 256, 0, stream>>>(acc1, b1, A2, 1024);

    // ================= conv2 =================
    gemm_wmma_bf16<<<dim3(NN / 16, 512 / 256), 128, 0, stream>>>(A2, W2T, h2, NN, 512, 1024);
    alpha_kernel<<<NN, 256, 0, stream>>>(h2, at_s2, at_d2, aS, aD, 64);

    fill_f32<<<nblk((size_t)NN * NHEAD, 256), 256, 0, stream>>>(mBuf, -3.0e38f, (size_t)NN * NHEAD);
    fill_f32<<<nblk((size_t)NN * NHEAD, 256), 256, 0, stream>>>(dnB, 0.f, (size_t)NN * NHEAD);
    fill_f32<<<nblk((size_t)NN * 512, 256), 256, 0, stream>>>(acc2, 0.f, (size_t)NN * 512);

    edge_pass1<<<nblk((size_t)EP * NHEAD, 256), 256, 0, stream>>>(ei, aS, aD, eb, mBuf);
    edge_pass2<<<nblk((size_t)EP * NHEAD, 256), 256, 0, stream>>>(ei, eb, mBuf, dnB);
    edge_pass3<<<EP, 256, 0, stream>>>(ei, eb, dnB, h2, acc2, 512, 64);
    finalize_f32<<<NN, 256, 0, stream>>>(acc2, b2, 512);

    // ================= pool + MLP head =================
    fill_f32<<<nblk((size_t)BGRAPH * 512, 256), 256, 0, stream>>>(pooled, 0.f, (size_t)BGRAPH * 512);
    fill_f32<<<1, 64, 0, stream>>>(counts, 0.f, (size_t)BGRAPH);
    pool_kernel<<<NN, 256, 0, stream>>>(acc2, batch, pooled, counts);
    fc_head<<<1, 512, 0, stream>>>(pooled, counts, fc1w, fc1b, fc2w, fc2b, out);
}
